// LSTM_model_37022618092270
// MI455X (gfx1250) — compile-verified, matrix operand order checked by reading, not run on hardware
//
#include <hip/hip_runtime.h>
#include <cstdint>
#include <cstddef>

typedef __attribute__((ext_vector_type(16))) __bf16 v16bf;
typedef __attribute__((ext_vector_type(8)))  float  v8f;

union FragB16 {
    v16bf v;
    uint4 q[2];
};

struct FragSet {
    FragB16 a, w0, w1, w2, w3;
};

static __device__ __forceinline__ uint16_t f2bf(float f) {
    union { float f; uint32_t u; } x; x.f = f;
    uint32_t u = x.u + 0x7FFFu + ((x.u >> 16) & 1u);  // round-to-nearest-even
    return (uint16_t)(u >> 16);
}

// ---------------- prep kernels ----------------

__global__ void k_f32_to_bf16(const float* __restrict__ src,
                              uint16_t* __restrict__ dst, int n) {
    int i = blockIdx.x * blockDim.x + threadIdx.x;
    if (i < n) dst[i] = f2bf(src[i]);
}

// Build Wcat^T: dst[n * 2048 + k] = (k < 1024 ? W[k,n] : U[k-1024,n]) as bf16.
// W,U are [1024,4096] row-major fp32. dst is [4096 n][2048 k] bf16 (N-major).
__global__ void k_build_wcat(const float* __restrict__ W,
                             const float* __restrict__ U,
                             uint16_t* __restrict__ dst) {
    int i = blockIdx.x * blockDim.x + threadIdx.x;  // 0 .. 4096*2048-1
    int k = i & 2047;
    int n = i >> 11;
    float v = (k < 1024) ? W[(size_t)k * 4096 + n]
                         : U[(size_t)(k - 1024) * 4096 + n];
    dst[i] = f2bf(v);
}

__global__ void k_zero_f32(float* __restrict__ p, int n) {
    int i = blockIdx.x * blockDim.x + threadIdx.x;
    if (i < n) p[i] = 0.0f;
}

__global__ void k_zero_u16(uint16_t* __restrict__ p, int n) {
    int i = blockIdx.x * blockDim.x + threadIdx.x;
    if (i < n) p[i] = 0;
}

// ---------------- fused LSTM step (one layer) ----------------
//
// z = [A0 | A1] @ WcatT   (M=64, K=2048, N=4096), then gate math in-register.
// Each wave: one 16-row M-tile x one 16-col n-tile x all 4 gates.
// All hot-loop loads are uniform-base + 32-bit lane offset (SADDR form) to
// minimize address VGPRs; fragments double-buffered (fA/fB) so the next
// iteration's 10 b128 loads overlap the current iteration's 4 WMMAs.
// grid = 64 (n-tiles), block = 128 (4 waves = 4 M-tiles).

// AKB: byte displacement into the A segment; WKB: byte displacement into Wcat row.
#define LOAD_SET(F, ABASE, AOFF, AKB, WKB)                                  \
    {                                                                       \
        const uint32_t akb = (uint32_t)(AKB) + h16;                         \
        const uint32_t wkb = (uint32_t)(WKB) + h16;                         \
        (F).a.q[0]  = *(const uint4*)((ABASE) + (AOFF) + akb);              \
        (F).a.q[1]  = *(const uint4*)((ABASE) + (AOFF) + akb + 32);         \
        (F).w0.q[0] = *(const uint4*)(wb + woff0 + wkb);                    \
        (F).w0.q[1] = *(const uint4*)(wb + woff0 + wkb + 32);               \
        (F).w1.q[0] = *(const uint4*)(wb + woff1 + wkb);                    \
        (F).w1.q[1] = *(const uint4*)(wb + woff1 + wkb + 32);               \
        (F).w2.q[0] = *(const uint4*)(wb + woff2 + wkb);                    \
        (F).w2.q[1] = *(const uint4*)(wb + woff2 + wkb + 32);               \
        (F).w3.q[0] = *(const uint4*)(wb + woff3 + wkb);                    \
        (F).w3.q[1] = *(const uint4*)(wb + woff3 + wkb + 32);               \
    }

#define MMA_SET(F)                                                          \
    acc0 = __builtin_amdgcn_wmma_f32_16x16x32_bf16(false, (F).a.v, false,   \
             (F).w0.v, (short)0, acc0, false, false);                       \
    acc1 = __builtin_amdgcn_wmma_f32_16x16x32_bf16(false, (F).a.v, false,   \
             (F).w1.v, (short)0, acc1, false, false);                       \
    acc2 = __builtin_amdgcn_wmma_f32_16x16x32_bf16(false, (F).a.v, false,   \
             (F).w2.v, (short)0, acc2, false, false);                       \
    acc3 = __builtin_amdgcn_wmma_f32_16x16x32_bf16(false, (F).a.v, false,   \
             (F).w3.v, (short)0, acc3, false, false);

__global__ __launch_bounds__(128)
void k_lstm_step(const uint16_t* __restrict__ A0, int a0_stride_bytes,
                 const uint16_t* __restrict__ A1,
                 const uint16_t* __restrict__ Wt,
                 float* __restrict__ Cst,        // [64][1024] in/out
                 float* __restrict__ Hf,         // [64][1024] out fp32
                 uint16_t* __restrict__ Hb)      // [64][1024] out bf16
{
    const int lane = threadIdx.x & 31;
    const int wave = threadIdx.x >> 5;   // 0..3 -> M-tile
    const int half = lane >> 4;          // 0 or 1
    const int idx  = lane & 15;
    const uint32_t h16 = (uint32_t)half * 16;   // byte offset of lane half
    const int m0   = wave * 16;
    const int n0   = blockIdx.x * 16;
    const int arow = m0 + idx;

    v8f acc0 = {}, acc1 = {}, acc2 = {}, acc3 = {};

    // Uniform bases (stay in SGPRs) + 32-bit per-lane byte offsets.
    const char* __restrict__ a0b = (const char*)A0;
    const char* __restrict__ a1b = (const char*)A1;
    const char* __restrict__ wb  = (const char*)Wt;
    const uint32_t aoff0 = (uint32_t)arow * (uint32_t)a0_stride_bytes;
    const uint32_t aoff1 = (uint32_t)arow * 2048u;              // 1024 bf16 row
    const uint32_t woff0 = (uint32_t)(0 * 1024 + n0 + idx) * 4096u; // 2048 bf16 row
    const uint32_t woff1 = (uint32_t)(1 * 1024 + n0 + idx) * 4096u;
    const uint32_t woff2 = (uint32_t)(2 * 1024 + n0 + idx) * 4096u;
    const uint32_t woff3 = (uint32_t)(3 * 1024 + n0 + idx) * 4096u;

    FragSet fA, fB;

    // ---- half 1: k in [0,1024), A from A0 (x_t or h1_new) ----
    LOAD_SET(fA, a0b, aoff0, 0, 0);
    for (int kt = 0; kt < 30; kt += 2) {
        LOAD_SET(fB, a0b, aoff0, (kt + 1) * 64, (kt + 1) * 64);
        MMA_SET(fA);
        LOAD_SET(fA, a0b, aoff0, (kt + 2) * 64, (kt + 2) * 64);
        MMA_SET(fB);
    }
    LOAD_SET(fB, a0b, aoff0, 31 * 64, 31 * 64);
    MMA_SET(fA);                              // iteration 30
    // ---- boundary: prefetch first h-half set before consuming iter 31 ----
    LOAD_SET(fA, a1b, aoff1, 0, 32 * 64);
    MMA_SET(fB);                              // iteration 31

    // ---- half 2: k in [1024,2048), A from A1 (h_old) ----
    for (int kt = 32; kt < 62; kt += 2) {
        LOAD_SET(fB, a1b, aoff1, (kt - 31) * 64, (kt + 1) * 64);
        MMA_SET(fA);
        LOAD_SET(fA, a1b, aoff1, (kt - 30) * 64, (kt + 2) * 64);
        MMA_SET(fB);
    }
    LOAD_SET(fB, a1b, aoff1, 31 * 64, 63 * 64);
    MMA_SET(fA);                              // iteration 62
    MMA_SET(fB);                              // iteration 63

    // In-register gate math: element e of accumulator, lane l -> (M = e + 8*half, N = idx)
    const uint32_t n = (uint32_t)(n0 + idx);
#pragma unroll
    for (int e = 0; e < 8; ++e) {
        const uint32_t bi = (uint32_t)(m0 + e) + (uint32_t)half * 8u; // batch row
        const uint32_t off = bi * 1024u + n;
        float zi = acc0[e], zf = acc1[e], zg = acc2[e], zo = acc3[e];
        float si = 1.0f / (1.0f + __expf(-zi));
        float sf = 1.0f / (1.0f + __expf(-zf));
        float so = 1.0f / (1.0f + __expf(-zo));
        float cn = sf * Cst[off] + si * tanhf(zg);
        float hn = so * tanhf(cn);
        Cst[off] = cn;
        Hf[off]  = hn;
        Hb[off]  = f2bf(hn);
    }
}

// ---------------- final Dense(1) ----------------
__global__ __launch_bounds__(256)
void k_dense(const float* __restrict__ H2, const float* __restrict__ Wd,
             const float* __restrict__ bd, float* __restrict__ y) {
    __shared__ float red[8];
    const int b = blockIdx.x;
    float s = 0.0f;
    for (int i = threadIdx.x; i < 1024; i += 256)
        s += H2[(size_t)b * 1024 + i] * Wd[i];
    for (int o = 16; o > 0; o >>= 1) s += __shfl_down(s, o, 32);
    if ((threadIdx.x & 31) == 0) red[threadIdx.x >> 5] = s;
    __syncthreads();
    if (threadIdx.x == 0) {
        float t = 0.0f;
        for (int w = 0; w < 8; ++w) t += red[w];
        y[b] = t + bd[0];
    }
}

// ---------------- host ----------------
extern "C" void kernel_launch(void* const* d_in, const int* in_sizes, int n_in,
                              void* d_out, int out_size, void* d_ws, size_t ws_size,
                              hipStream_t stream) {
    (void)in_sizes; (void)n_in; (void)out_size; (void)ws_size;
    const float* x  = (const float*)d_in[0];
    const float* W1 = (const float*)d_in[1];
    const float* U1 = (const float*)d_in[2];
    const float* W2 = (const float*)d_in[3];
    const float* U2 = (const float*)d_in[4];
    const float* Wd = (const float*)d_in[5];
    const float* bd = (const float*)d_in[6];

    constexpr int B = 64, T = 256, D = 1024, N = 1024;

    float* out = (float*)d_out;
    float* y  = out;
    float* h1 = out + B;            // 64 y values then states
    float* c1 = h1 + B * N;
    float* h2 = c1 + B * N;
    float* c2 = h2 + B * N;

    uint8_t* wsp = (uint8_t*)d_ws;
    auto carve = [&](size_t bytes) {
        void* p = (void*)wsp;
        wsp += (bytes + 255) & ~(size_t)255;
        return p;
    };
    uint16_t* Wc1 = (uint16_t*)carve((size_t)4096 * 2048 * 2);  // 16 MB
    uint16_t* Wc2 = (uint16_t*)carve((size_t)4096 * 2048 * 2);  // 16 MB
    uint16_t* Xb  = (uint16_t*)carve((size_t)B * T * D * 2);    // 32 MB
    uint16_t* h1b = (uint16_t*)carve((size_t)2 * B * N * 2);    // ping-pong
    uint16_t* h2b = (uint16_t*)carve((size_t)2 * B * N * 2);    // ping-pong

    // Prep: bf16 x, transposed+concatenated bf16 weights, zero states.
    k_f32_to_bf16<<<(B * T * D) / 256, 256, 0, stream>>>(x, Xb, B * T * D);
    k_build_wcat<<<(4096 * 2048) / 256, 256, 0, stream>>>(W1, U1, Wc1);
    k_build_wcat<<<(4096 * 2048) / 256, 256, 0, stream>>>(W2, U2, Wc2);
    k_zero_f32<<<(4 * B * N) / 256, 256, 0, stream>>>(h1, 4 * B * N); // h1,c1,h2,c2
    k_zero_u16<<<(4 * B * N) / 256, 256, 0, stream>>>(h1b, 4 * B * N); // h1b+h2b (contiguous)

    // Sequential recurrence: 2 fused GEMM+gate kernels per step.
    for (int t = 0; t < T; ++t) {
        const int p = t & 1;
        // Layer 1: A = [x_t | h1_old]
        k_lstm_step<<<64, 128, 0, stream>>>(
            Xb + (size_t)t * D, T * D * 2 /* row stride bytes */,
            h1b + (size_t)p * B * N,
            Wc1, c1, h1,
            h1b + (size_t)(1 - p) * B * N);
        // Layer 2: A = [h1_new | h2_old]
        k_lstm_step<<<64, 128, 0, stream>>>(
            h1b + (size_t)(1 - p) * B * N, N * 2 /* row stride bytes */,
            h2b + (size_t)p * B * N,
            Wc2, c2, h2,
            h2b + (size_t)(1 - p) * B * N);
    }

    // y = h2 @ Wd + bd
    k_dense<<<B, 256, 0, stream>>>(h2, Wd, bd, y);
}
